// LAAT_28140625723738
// MI455X (gfx1250) — matrix operand order repaired and checked by reading, not compiled, not run on hardware
//
#include <hip/hip_runtime.h>
#include <math.h>

#define DIM    2048
#define NH     16
#define QRANK  768
#define KVRANK 512
#define QKS    128
#define QKR    64
#define QKT    192
#define VDIM   128
#define BSZ    2
#define SEQL   1024
#define ROWS   (BSZ*SEQL)

typedef float    v8f   __attribute__((ext_vector_type(8)));
typedef __bf16   bf8   __attribute__((ext_vector_type(8)));
typedef __bf16   bf16v __attribute__((ext_vector_type(16)));
typedef unsigned u32x4 __attribute__((ext_vector_type(4)));
typedef unsigned u32x8 __attribute__((ext_vector_type(8)));

union Frag { bf16v v; bf8 h[2]; };

__device__ inline __bf16 f2bf(float f) {
  unsigned u = __builtin_bit_cast(unsigned, f);
  unsigned r = u + 0x7FFFu + ((u >> 16) & 1u);
  unsigned short s = (unsigned short)(r >> 16);
  return __builtin_bit_cast(__bf16, s);
}

__device__ inline v8f wmma_bf16(bf16v a, bf16v b, v8f c) {
  return __builtin_amdgcn_wmma_f32_16x16x32_bf16(false, a, false, b, (short)0, c, false, false);
}

// --- CDNA5 data movers -----------------------------------------------------
// Tensor Data Mover: load a 2-D tile (rows x 32 bf16) with source row stride
// `stride_halfs` into LDS rows padded to 40 halfs (pad 4 dwords every 16).
// D# bit layout per CDNA5 ISA.
__device__ inline void tdm_load_tile32(unsigned lds_off, const void* gsrc,
                                       unsigned rows, unsigned stride_halfs) {
  unsigned long long ga = (unsigned long long)gsrc;
  u32x4 g0;
  g0[0] = 1u;                                           // count=1, user D#
  g0[1] = lds_off;                                      // lds_addr (bytes)
  g0[2] = (unsigned)ga;                                 // global_addr[31:0]
  g0[3] = ((unsigned)(ga >> 32) & 0x01FFFFFFu)          // global_addr[56:32]
          | 0x80000000u;                                // type = 2 ("image")
  u32x8 g1;
  g1[0] = (1u << 16)        // data_size = 1 (2 bytes)
        | (1u << 20)        // pad_enable
        | (3u << 22)        // pad_interval: every 16 dwords (=32 halfs)
        | (3u << 25);       // pad_amount: 4 dwords (=8 halfs) -> 40-half rows
  g1[1] = 32u << 16;        // tensor_dim0[15:0] = 32 halfs per row
  g1[2] = rows << 16;       // tensor_dim0[31:16]=0 | tensor_dim1[15:0] = rows
  g1[3] = 32u << 16;        // tensor_dim1[31:16]=0 | tile_dim0 = 32
  g1[4] = rows;             // tile_dim1 = rows | tile_dim2 = 0
  g1[5] = stride_halfs;     // tensor_dim0_stride[31:0] (element units)
  g1[6] = 0u;               // stride0[47:32] | tensor_dim1_stride[15:0]
  g1[7] = 0u;
  asm volatile("tensor_load_to_lds %0, %1" :: "s"(g0), "s"(g1) : "memory");
}

// Async global -> LDS copy: 16 bytes per lane, tracked with ASYNCcnt.
__device__ inline void async_g2l_b128(void* lds, const void* g) {
  unsigned l = (unsigned)(unsigned long long)lds;
  asm volatile("global_load_async_to_lds_b128 %0, %1, off"
               :: "v"(l), "v"(g) : "memory");
}
__device__ inline void wait_async0() {
  asm volatile("s_wait_asynccnt 0x0" ::: "memory");
}
__device__ inline void wait_async4() {   // older batch of 4 complete (in-order)
  asm volatile("s_wait_asynccnt 0x4" ::: "memory");
}

// ---------------------------------------------------------------------------
// Generic GEMM: C(MxN) = A(MxK) * W(NxK)^T + bias, bf16 in, f32 or bf16 out.
// Block tile 256x64, 8 waves (4 along M x 2 along N), wave tile 64x32.
// Double-buffered staging: A tile via TDM (wave 0, up to 2 descriptors in
// flight, s_wait_tensorcnt 1), W tile via async-to-LDS (waves 0-1, up to 8
// b128s in flight, s_wait_asynccnt 4) so DMA of tile k+1 overlaps WMMA on k.
// ---------------------------------------------------------------------------
template <bool OBF>
__global__ __launch_bounds__(256) void gemm_k(
    const __bf16* __restrict__ A, long sAb, int lda,
    const __bf16* __restrict__ W, long sWb, int ldw,
    void* __restrict__ Cv, long sCb, int ldc,
    const float* __restrict__ bias, int M, int N, int K)
{
  __shared__ __bf16 As[2][256][40];
  __shared__ __bf16 Ws[2][64][40];
  const int tid  = threadIdx.x;
  const int wid  = tid >> 5;
  const int lane = tid & 31;
  const int l16  = lane & 15;
  const int ah   = (lane < 16) ? 0 : 8;   // A-fragment half offset
  const int bh   = (lane < 16) ? 0 : 8;   // B-fragment half offset
  const int m0   = blockIdx.x * 256;
  const int n0   = blockIdx.y * 64;
  A += (long)blockIdx.z * sAb;
  W += (long)blockIdx.z * sWb;
  const int wm = (wid & 3) * 64;
  const int wn = (wid >> 2) * 32;
  const unsigned as_off[2] = {
      (unsigned)(unsigned long long)(void*)&As[0][0][0],
      (unsigned)(unsigned long long)(void*)&As[1][0][0] };
  const __bf16* Arow = A + (long)m0 * lda;
  const __bf16* Wrow = W + (long)(n0 + (tid & 63)) * ldw;
  const int ksteps = K >> 5;
  v8f acc[4][2] = {};
  // prologue: stage tile 0 into buffer 0
  if (wid == 0)
    tdm_load_tile32(as_off[0], Arow, 256u, (unsigned)lda);
  if (tid < 64) {
    async_g2l_b128(&Ws[0][tid][0],  Wrow);
    async_g2l_b128(&Ws[0][tid][8],  Wrow + 8);
    async_g2l_b128(&Ws[0][tid][16], Wrow + 16);
    async_g2l_b128(&Ws[0][tid][24], Wrow + 24);
  }
  for (int ks = 0; ks < ksteps; ++ks) {
    const int cur  = ks & 1;
    const int nxt  = cur ^ 1;
    const bool more = (ks + 1) < ksteps;
    const int k1 = (ks + 1) << 5;
    if (more) { // kick off DMA of the next tile into the other buffer
      if (wid == 0)
        tdm_load_tile32(as_off[nxt], Arow + k1, 256u, (unsigned)lda);
      if (tid < 64) {
        async_g2l_b128(&Ws[nxt][tid][0],  Wrow + k1);
        async_g2l_b128(&Ws[nxt][tid][8],  Wrow + k1 + 8);
        async_g2l_b128(&Ws[nxt][tid][16], Wrow + k1 + 16);
        async_g2l_b128(&Ws[nxt][tid][24], Wrow + k1 + 24);
      }
    }
    // wait for the *current* tile only (in-order completion)
    if (tid < 64) { if (more) wait_async4(); else wait_async0(); }
    if (wid == 0) {
      if (more) __builtin_amdgcn_s_wait_tensorcnt(1);
      else      __builtin_amdgcn_s_wait_tensorcnt(0);
    }
    __syncthreads();
    Frag wf[2];
#pragma unroll
    for (int j = 0; j < 2; ++j) {
      const __bf16* p = &Ws[cur][wn + j * 16 + l16][0];
      wf[j].h[0] = *(const bf8*)(p + bh);
      wf[j].h[1] = *(const bf8*)(p + bh + 16);
    }
#pragma unroll
    for (int i = 0; i < 4; ++i) {
      Frag af;
      const __bf16* p = &As[cur][wm + i * 16 + l16][0];
      af.h[0] = *(const bf8*)(p + ah);
      af.h[1] = *(const bf8*)(p + ah + 16);
#pragma unroll
      for (int j = 0; j < 2; ++j)
        acc[i][j] = wmma_bf16(af.v, wf[j].v, acc[i][j]);
    }
    __syncthreads();  // all reads of buf `cur` done before it is refilled
  }
  const int roff = (lane < 16) ? 0 : 8;
#pragma unroll
  for (int i = 0; i < 4; ++i)
#pragma unroll
    for (int j = 0; j < 2; ++j) {
      const int col = n0 + wn + j * 16 + l16;
      const float bv = bias ? bias[col] : 0.f;
#pragma unroll
      for (int r = 0; r < 8; ++r) {
        const int row = m0 + wm + i * 16 + r + roff;
        const float v = acc[i][j][r] + bv;
        if constexpr (OBF)
          ((__bf16*)Cv)[(long)blockIdx.z * sCb + (long)row * ldc + col] = f2bf(v);
        else
          ((float*)Cv)[(long)blockIdx.z * sCb + (long)row * ldc + col] = v;
      }
    }
}

// ---------------------------------------------------------------------------
// Flash attention core. Grid (SEQL/128, NH, BSZ), 256 threads (8 waves).
// Each wave owns 16 query rows; keys processed 32 at a time.
// ---------------------------------------------------------------------------
__global__ __launch_bounds__(256) void flash_k(
    const __bf16* __restrict__ qp,   // (B,S,H,512)
    const __bf16* __restrict__ qr,   // (B,S,H,64)
    const __bf16* __restrict__ kvn,  // (B,S,512)
    const __bf16* __restrict__ kr,   // (B,S,64)
    __bf16* __restrict__ ctx)        // (B,S,H,512)
{
  __shared__ __bf16 Vt[512][40];       // transposed V tile: Vt[c][key]
  __shared__ __bf16 Pst[8][16][40];    // per-wave P staging (C->A layout)
  const int tid  = threadIdx.x;
  const int wid  = tid >> 5;
  const int lane = tid & 31;
  const int l16  = lane & 15;
  const int ah   = (lane < 16) ? 0 : 8;
  const int bh   = (lane < 16) ? 0 : 8;
  const int bb   = blockIdx.z;
  const int hh   = blockIdx.y;
  const int q0   = blockIdx.x * 128;
  const int qw   = q0 + wid * 16;
  const long qrow = (long)(bb * SEQL + qw + l16) * NH + hh;
  const __bf16* qpL = qp + qrow * KVRANK;
  const __bf16* qrL = qr + qrow * QKR;
  v8f O[32];
#pragma unroll
  for (int i = 0; i < 32; ++i) O[i] = (v8f)0.f;
  float mrow[8], lrow[8];
#pragma unroll
  for (int j = 0; j < 8; ++j) { mrow[j] = -1e30f; lrow[j] = 0.f; }
  const int roff = (lane < 16) ? 0 : 8;
  const float scale = 0.0721687836487f;   // 1/sqrt(192)
  const int nkt = q0 / 32 + 4;
  for (int kt = 0; kt < nkt; ++kt) {
    const int kb = kt * 32;
    __syncthreads();
    { // cooperative transpose of 32x512 V tile into LDS
      const __bf16* src = kvn + (long)(bb * SEQL + kb) * KVRANK;
#pragma unroll
      for (int i = 0; i < 8; ++i) {
        const int c = tid * 8 + i;        // 0..2047 chunks of 8 halfs
        const int krw = c >> 6;           // 64 chunks per key row
        const int cof = (c & 63) * 8;
        bf8 d = *(const bf8*)(src + (long)krw * KVRANK + cof);
#pragma unroll
        for (int e = 0; e < 8; ++e) Vt[cof + e][krw] = d[e];
      }
    }
    __syncthreads();
    if (kb <= qw + 15) {
      v8f S0 = (v8f)0.f, S1 = (v8f)0.f;
      const __bf16* k0p = kvn + (long)(bb * SEQL + kb + l16) * KVRANK;
      const __bf16* k1p = k0p + 16 * KVRANK;
      const __bf16* r0p = kr + (long)(bb * SEQL + kb + l16) * QKR;
      const __bf16* r1p = r0p + 16 * QKR;
#pragma unroll
      for (int ks = 0; ks < 18; ++ks) {
        Frag qf, b0, b1;
        if (ks < 16) {
          const int kc = ks * 32;
          qf.h[0] = *(const bf8*)(qpL + kc + ah);
          qf.h[1] = *(const bf8*)(qpL + kc + 16 + ah);
          b0.h[0] = *(const bf8*)(k0p + kc + bh);
          b0.h[1] = *(const bf8*)(k0p + kc + bh + 16);
          b1.h[0] = *(const bf8*)(k1p + kc + bh);
          b1.h[1] = *(const bf8*)(k1p + kc + bh + 16);
        } else {
          const int kc = (ks - 16) * 32;
          qf.h[0] = *(const bf8*)(qrL + kc + ah);
          qf.h[1] = *(const bf8*)(qrL + kc + 16 + ah);
          b0.h[0] = *(const bf8*)(r0p + kc + bh);
          b0.h[1] = *(const bf8*)(r0p + kc + bh + 16);
          b1.h[0] = *(const bf8*)(r1p + kc + bh);
          b1.h[1] = *(const bf8*)(r1p + kc + bh + 16);
        }
        S0 = wmma_bf16(qf.v, b0.v, S0);
        S1 = wmma_bf16(qf.v, b1.v, S1);
      }
      float alpha[8];
#pragma unroll
      for (int j = 0; j < 8; ++j) {
        const int grow = qw + j + roff;
        const int key0 = kb + l16;
        float s0 = S0[j] * scale + (key0 > grow ? -1e9f : 0.f);
        float s1 = S1[j] * scale + (key0 + 16 > grow ? -1e9f : 0.f);
        float mt = fmaxf(s0, s1);
        mt = fmaxf(mt, __shfl_xor(mt, 1, 32));
        mt = fmaxf(mt, __shfl_xor(mt, 2, 32));
        mt = fmaxf(mt, __shfl_xor(mt, 4, 32));
        mt = fmaxf(mt, __shfl_xor(mt, 8, 32));
        const float mn = fmaxf(mrow[j], mt);
        const float a  = __expf(mrow[j] - mn);
        const float p0 = __expf(s0 - mn);
        const float p1 = __expf(s1 - mn);
        float ls = p0 + p1;
        ls += __shfl_xor(ls, 1, 32);
        ls += __shfl_xor(ls, 2, 32);
        ls += __shfl_xor(ls, 4, 32);
        ls += __shfl_xor(ls, 8, 32);
        mrow[j] = mn;
        lrow[j] = lrow[j] * a + ls;
        alpha[j] = a;
        Pst[wid][j + roff][l16]      = f2bf(p0);
        Pst[wid][j + roff][16 + l16] = f2bf(p1);
      }
      Frag pf;
      {
        const __bf16* p = &Pst[wid][l16][0];
        pf.h[0] = *(const bf8*)(p + ah);
        pf.h[1] = *(const bf8*)(p + ah + 16);
      }
#pragma unroll
      for (int nt = 0; nt < 32; ++nt) {
        Frag vf;
        const __bf16* p = &Vt[nt * 16 + l16][0];
        vf.h[0] = *(const bf8*)(p + bh);
        vf.h[1] = *(const bf8*)(p + bh + 16);
        v8f o = O[nt];
#pragma unroll
        for (int j = 0; j < 8; ++j) o[j] *= alpha[j];
        O[nt] = wmma_bf16(pf.v, vf.v, o);
      }
    }
  }
  float inv[8];
#pragma unroll
  for (int j = 0; j < 8; ++j) inv[j] = 1.f / lrow[j];
#pragma unroll
  for (int nt = 0; nt < 32; ++nt) {
#pragma unroll
    for (int j = 0; j < 8; ++j) {
      const long row = (long)bb * SEQL + qw + j + roff;
      ctx[(row * NH + hh) * (long)KVRANK + nt * 16 + l16] = f2bf(O[nt][j] * inv[j]);
    }
  }
}

// ---------------------------------------------------------------------------
// Elementwise / norm / rope helpers
// ---------------------------------------------------------------------------
__global__ void cvt_k(const float* __restrict__ in, __bf16* __restrict__ out, long n) {
  long i = (long)blockIdx.x * blockDim.x + threadIdx.x;
  const long stride = (long)gridDim.x * blockDim.x;
  for (; i < n; i += stride) out[i] = f2bf(in[i]);
}

__global__ void wkt_k(const float* __restrict__ kvup, __bf16* __restrict__ wkT) {
  const long t = (long)blockIdx.x * blockDim.x + threadIdx.x;
  if (t >= (long)NH * KVRANK * QKS) return;
  const int d = (int)(t & (QKS - 1));
  const int c = (int)((t >> 7) & (KVRANK - 1));
  const int h = (int)(t >> 16);
  wkT[t] = f2bf(kvup[((long)h * QKS + d) * KVRANK + c]);
}

__global__ __launch_bounds__(256) void rmsnorm_k(const float* __restrict__ in,
    const float* __restrict__ sc, __bf16* __restrict__ out, int N) {
  const int row = blockIdx.x;
  in  += (long)row * N;
  out += (long)row * N;
  float ss = 0.f;
  for (int i = threadIdx.x; i < N; i += 256) { const float v = in[i]; ss += v * v; }
  ss += __shfl_xor(ss, 1, 32);
  ss += __shfl_xor(ss, 2, 32);
  ss += __shfl_xor(ss, 4, 32);
  ss += __shfl_xor(ss, 8, 32);
  ss += __shfl_xor(ss, 16, 32);
  __shared__ float red[8];
  const int wid = threadIdx.x >> 5, lane = threadIdx.x & 31;
  if (lane == 0) red[wid] = ss;
  __syncthreads();
  float tot = 0.f;
#pragma unroll
  for (int i = 0; i < 8; ++i) tot += red[i];
  const float inv = rsqrtf(tot / (float)N + 1e-6f);
  for (int i = threadIdx.x; i < N; i += 256) out[i] = f2bf(in[i] * inv * sc[i]);
}

__global__ __launch_bounds__(128) void kvpost_k(const float* __restrict__ kv,
    const float* __restrict__ sc, __bf16* __restrict__ kvn, __bf16* __restrict__ krot) {
  const int row = blockIdx.x;
  const float* r = kv + (long)row * (KVRANK + QKR);
  float ss = 0.f;
  for (int i = threadIdx.x; i < KVRANK; i += 128) { const float v = r[i]; ss += v * v; }
  ss += __shfl_xor(ss, 1, 32);
  ss += __shfl_xor(ss, 2, 32);
  ss += __shfl_xor(ss, 4, 32);
  ss += __shfl_xor(ss, 8, 32);
  ss += __shfl_xor(ss, 16, 32);
  __shared__ float red[4];
  const int wid = threadIdx.x >> 5, lane = threadIdx.x & 31;
  if (lane == 0) red[wid] = ss;
  __syncthreads();
  const float tot = red[0] + red[1] + red[2] + red[3];
  const float inv = rsqrtf(tot / (float)KVRANK + 1e-6f);
  for (int i = threadIdx.x; i < KVRANK; i += 128)
    kvn[(long)row * KVRANK + i] = f2bf(r[i] * inv * sc[i]);
  if (threadIdx.x < QKR) {
    const int j = threadIdx.x, jj = j & 31;
    const float invf = __powf(10000.f, -(2.f * jj) / 64.f);
    const float ang = (float)(row & (SEQL - 1)) * invf;
    const float x = r[KVRANK + j];
    const float hx = (j < 32) ? -r[KVRANK + j + 32] : r[KVRANK + j - 32];
    krot[(long)row * QKR + j] = f2bf(x * __cosf(ang) + hx * __sinf(ang));
  }
}

__global__ void qpost_k(const float* __restrict__ qf,
                        __bf16* __restrict__ qs, __bf16* __restrict__ qrot) {
  const long t = (long)blockIdx.x * blockDim.x + threadIdx.x;
  if (t >= (long)ROWS * NH * QKR) return;
  const int j = (int)(t & 63);
  const int h = (int)((t >> 6) & 15);
  const long row = t >> 10;
  const float* base = qf + row * (NH * QKT) + h * QKT;
  const long so = (row * NH + h) * QKS;
  qs[so + j]      = f2bf(base[j]);
  qs[so + j + 64] = f2bf(base[j + 64]);
  const int jj = j & 31;
  const float invf = __powf(10000.f, -(2.f * jj) / 64.f);
  const float ang = (float)(row & (SEQL - 1)) * invf;
  const float x = base[QKS + j];
  const float hx = (j < 32) ? -base[QKS + j + 32] : base[QKS + j - 32];
  qrot[(row * NH + h) * QKR + j] = f2bf(x * __cosf(ang) + hx * __sinf(ang));
}

// ---------------------------------------------------------------------------
// Workspace layout (bytes). Buffers reused once their producers/consumers
// are done; high-water ~113.5 MiB.
// ---------------------------------------------------------------------------
static constexpr size_t OFF_XBF   = 0;                        // 8388608
static constexpr size_t OFF_WQD   = OFF_XBF   + 8388608;      // 3145728
static constexpr size_t OFF_WQU   = OFF_WQD   + 3145728;      // 4718592
static constexpr size_t OFF_WKVD  = OFF_WQU   + 4718592;      // 2359296
static constexpr size_t OFF_WKT   = OFF_WKVD  + 2359296;      // 2097152
static constexpr size_t OFF_WV    = OFF_WKT   + 2097152;      // 2097152
static constexpr size_t OFF_WOUT  = OFF_WV    + 2097152;      // 8388608
static constexpr size_t OFF_QLAT  = OFF_WOUT  + 8388608;      // f32 6291456
static constexpr size_t OFF_QNORM = OFF_QLAT  + 6291456;      // 3145728
static constexpr size_t OFF_QFULL = OFF_QNORM + 3145728;      // f32 25165824
static constexpr size_t OFF_QSTAT = OFF_QFULL + 25165824;     // 8388608
static constexpr size_t OFF_QROT  = OFF_QSTAT + 8388608;      // 4194304
static constexpr size_t OFF_KV    = OFF_QROT  + 4194304;      // f32 4718592
static constexpr size_t OFF_KVN   = OFF_KV    + 4718592;      // 2097152
static constexpr size_t OFF_KROT  = OFF_KVN   + 2097152;      // 262144
static constexpr size_t OFF_CTXC  = OFF_KROT  + 262144;       // 33554432
static constexpr size_t OFF_QPROJ = OFF_QLAT;                 // reuse (33.5MB fits)
static constexpr size_t OFF_CTX   = OFF_QSTAT;                // reuse (8.4MB)

extern "C" void kernel_launch(void* const* d_in, const int* in_sizes, int n_in,
                              void* d_out, int out_size, void* d_ws, size_t ws_size,
                              hipStream_t stream)
{
  const float* x         = (const float*)d_in[0];
  // d_in[1] = start_pos (0), d_in[2] = mask (causal, computed inline)
  const float* q_down_w  = (const float*)d_in[3];
  const float* q_down_b  = (const float*)d_in[4];
  const float* q_norm_s  = (const float*)d_in[5];
  const float* q_up_w    = (const float*)d_in[6];
  const float* q_up_b    = (const float*)d_in[7];
  const float* kv_down_w = (const float*)d_in[8];
  const float* kv_down_b = (const float*)d_in[9];
  const float* kv_norm_s = (const float*)d_in[10];
  const float* kv_up_w   = (const float*)d_in[11];
  const float* out_w     = (const float*)d_in[12];
  const float* out_b     = (const float*)d_in[13];

  char* ws = (char*)d_ws;
  auto BF  = [&](size_t off) { return (__bf16*)(ws + off); };
  auto F32 = [&](size_t off) { return (float*)(ws + off); };

  // --- precision conversion of activations and weights to bf16 ---
  cvt_k<<<2048, 256, 0, stream>>>(x,         BF(OFF_XBF),  (long)ROWS * DIM);
  cvt_k<<<768,  256, 0, stream>>>(q_down_w,  BF(OFF_WQD),  (long)QRANK * DIM);
  cvt_k<<<1024, 256, 0, stream>>>(q_up_w,    BF(OFF_WQU),  (long)NH * QKT * QRANK);
  cvt_k<<<576,  256, 0, stream>>>(kv_down_w, BF(OFF_WKVD), (long)(KVRANK + QKR) * DIM);
  cvt_k<<<512,  256, 0, stream>>>(kv_up_w + (long)NH * QKS * KVRANK,
                                  BF(OFF_WV), (long)NH * VDIM * KVRANK);
  cvt_k<<<2048, 256, 0, stream>>>(out_w,     BF(OFF_WOUT), (long)DIM * NH * VDIM);
  wkt_k<<<(NH * KVRANK * QKS + 255) / 256, 256, 0, stream>>>(kv_up_w, BF(OFF_WKT));

  // --- q path: down-proj, rmsnorm, up-proj, split + rope ---
  gemm_k<false><<<dim3(ROWS / 256, QRANK / 64, 1), 256, 0, stream>>>(
      BF(OFF_XBF), 0, DIM, BF(OFF_WQD), 0, DIM,
      F32(OFF_QLAT), 0, QRANK, q_down_b, ROWS, QRANK, DIM);
  rmsnorm_k<<<ROWS, 256, 0, stream>>>(F32(OFF_QLAT), q_norm_s, BF(OFF_QNORM), QRANK);
  gemm_k<false><<<dim3(ROWS / 256, (NH * QKT) / 64, 1), 256, 0, stream>>>(
      BF(OFF_QNORM), 0, QRANK, BF(OFF_WQU), 0, QRANK,
      F32(OFF_QFULL), 0, NH * QKT, q_up_b, ROWS, NH * QKT, QRANK);
  qpost_k<<<(ROWS * NH * QKR + 255) / 256, 256, 0, stream>>>(
      F32(OFF_QFULL), BF(OFF_QSTAT), BF(OFF_QROT));

  // --- kv path: down-proj, rmsnorm + rope ---
  gemm_k<false><<<dim3(ROWS / 256, (KVRANK + QKR) / 64, 1), 256, 0, stream>>>(
      BF(OFF_XBF), 0, DIM, BF(OFF_WKVD), 0, DIM,
      F32(OFF_KV), 0, KVRANK + QKR, kv_down_b, ROWS, KVRANK + QKR, DIM);
  kvpost_k<<<ROWS, 128, 0, stream>>>(F32(OFF_KV), kv_norm_s, BF(OFF_KVN), BF(OFF_KROT));

  // --- q_static -> latent projection, batched over heads ---
  gemm_k<true><<<dim3(ROWS / 256, KVRANK / 64, NH), 256, 0, stream>>>(
      BF(OFF_QSTAT), QKS, NH * QKS,
      BF(OFF_WKT), (long)KVRANK * QKS, QKS,
      BF(OFF_QPROJ), KVRANK, NH * KVRANK, nullptr, ROWS, KVRANK, QKS);

  // --- flash attention over the latent + rope dims ---
  flash_k<<<dim3(SEQL / 128, NH, BSZ), 256, 0, stream>>>(
      BF(OFF_QPROJ), BF(OFF_QROT), BF(OFF_KVN), BF(OFF_KROT), BF(OFF_CTXC));

  // --- latent ctx -> v-dim projection, batched over heads ---
  gemm_k<true><<<dim3(ROWS / 256, VDIM / 64, NH), 256, 0, stream>>>(
      BF(OFF_CTXC), KVRANK, NH * KVRANK,
      BF(OFF_WV), (long)VDIM * KVRANK, KVRANK,
      BF(OFF_CTX), VDIM, NH * VDIM, nullptr, ROWS, VDIM, KVRANK);

  // --- output projection, f32 directly into d_out ---
  gemm_k<false><<<dim3(ROWS / 256, DIM / 64, 1), 256, 0, stream>>>(
      BF(OFF_CTX), 0, NH * VDIM, BF(OFF_WOUT), 0, NH * VDIM,
      (float*)d_out, 0, DIM, out_b, ROWS, DIM, NH * VDIM);

  (void)in_sizes; (void)n_in; (void)out_size; (void)ws_size;
}
